// QuantumGenerator_88983132439015
// MI455X (gfx1250) — compile-verified, matrix operand order checked by reading, not run on hardware
//
#include <hip/hip_runtime.h>
#include <hip/hip_bf16.h>

typedef __attribute__((ext_vector_type(16))) _Float16 v16h;
typedef __attribute__((ext_vector_type(8)))  float    v8f;

#define NQ  5
#define DIM 32

__device__ __forceinline__ void cmulf(float ar, float ai, float br, float bi,
                                      float& cr, float& ci) {
    cr = ar * br - ai * bi;
    ci = ar * bi + ai * br;
}

// Apply a 2x2 complex gate on qubit q to one 32-amplitude column (lane-private LDS row).
__device__ __forceinline__ void apply_1q(float* re, float* im, int q,
                                         float u00r, float u00i, float u01r, float u01i,
                                         float u10r, float u10i, float u11r, float u11i) {
    const int shift = 1 << (4 - q);
    const int mask  = shift - 1;
    #pragma unroll
    for (int kk = 0; kk < 16; ++kk) {
        int k0 = ((kk & ~mask) << 1) | (kk & mask);
        int k1 = k0 | shift;
        float s0r = re[k0], s0i = im[k0];
        float s1r = re[k1], s1i = im[k1];
        re[k0] = u00r * s0r - u00i * s0i + u01r * s1r - u01i * s1i;
        im[k0] = u00r * s0i + u00i * s0r + u01r * s1i + u01i * s1r;
        re[k1] = u10r * s0r - u10i * s0i + u11r * s1r - u11i * s1i;
        im[k1] = u10r * s0i + u10i * s0r + u11r * s1i + u11i * s1r;
    }
}

// One wave: lane c evolves basis column c of the circuit unitary, then writes
// six WMMA-ready f16 B-fragments (Re, Im, -Im for N-tiles 0 and 1) into d_ws.
// B-fragment layout (16-bit B, 32x16, K-major per lane, mirrors ISA A layout):
//   lane<16 : N=lane,    K = {0..7,16..23}  packed as V0..V7 (2 halves each)
//   lane>=16: N=lane-16, K = {8..15,24..31}
__global__ __launch_bounds__(32) void build_unitary(const float* __restrict__ p,
                                                    _Float16* __restrict__ frag) {
    __shared__ float sRe[DIM][DIM];  // [column c][amplitude n]
    __shared__ float sIm[DIM][DIM];
    const int c = threadIdx.x;
    float* re = &sRe[c][0];
    float* im = &sIm[c][0];
    #pragma unroll
    for (int n = 0; n < DIM; ++n) { re[n] = (n == c) ? 1.f : 0.f; im[n] = 0.f; }

    int idx = NQ;
    for (int layer = 0; layer < 4; ++layer) {
        for (int q = 0; q < NQ; ++q) {
            float a = p[idx], b = p[idx + 1], g = p[idx + 2];
            idx += 3;
            float ca, sa, cb, sb, cg, sg;
            __sincosf(0.5f * a, &sa, &ca);
            __sincosf(0.5f * b, &sb, &cb);
            __sincosf(0.5f * g, &sg, &cg);
            // U = Rz(g) @ Ry(b) @ Rz(a)
            float tr, ti;
            cmulf(cg, -sg, ca, -sa, tr, ti); float u00r = tr * cb, u00i = ti * cb;
            cmulf(cg, -sg, ca,  sa, tr, ti); float u01r = -tr * sb, u01i = -ti * sb;
            cmulf(cg,  sg, ca, -sa, tr, ti); float u10r = tr * sb, u10i = ti * sb;
            cmulf(cg,  sg, ca,  sa, tr, ti); float u11r = tr * cb, u11i = ti * cb;
            apply_1q(re, im, q, u00r, u00i, u01r, u01i, u10r, u10i, u11r, u11i);
        }
        // CNOT ring: ctrl q -> target (q + layer+1) % 5. Permutation is an
        // involution: swap pairs (k, k|tbit) where ctrl bit set, tbit clear.
        int r = layer + 1;
        for (int q = 0; q < NQ; ++q) {
            int t = (q + r) % NQ;
            int cbit = 1 << (4 - q);
            int tbit = 1 << (4 - t);
            #pragma unroll
            for (int k = 0; k < DIM; ++k) {
                if ((k & cbit) && !(k & tbit)) {
                    int k2 = k | tbit;
                    float tmp;
                    tmp = re[k]; re[k] = re[k2]; re[k2] = tmp;
                    tmp = im[k]; im[k] = im[k2]; im[k2] = tmp;
                }
            }
        }
    }
    for (int q = 0; q < NQ; ++q) {
        float t0 = p[idx], t1 = p[idx + 1];
        idx += 2;
        float c0, s0, c1, s1;
        __sincosf(0.5f * t0, &s0, &c0);
        __sincosf(0.5f * t1, &s1, &c1);
        apply_1q(re, im, q, c0, 0.f, 0.f, -s0, 0.f, -s0, c0, 0.f);   // Rx
        apply_1q(re, im, q, c1, 0.f, -s1, 0.f, s1, 0.f, c1, 0.f);    // Ry
    }

    // Fold batch-uniform base phase exp(-i/2 * p[:5]@ZS[:,c]) and 1/sqrt(32)
    // into column c:  M[c][n] = U[n][c] * phase0_c / sqrt(32).
    float base = 0.f;
    #pragma unroll
    for (int q = 0; q < NQ; ++q)
        base += ((c >> (4 - q)) & 1) ? -p[q] : p[q];
    float es, ec;
    __sincosf(0.5f * base, &es, &ec);
    const float inv = 0.1767766952966369f;  // 1/sqrt(32)
    float scr = ec * inv, sci = -es * inv;
    #pragma unroll
    for (int n = 0; n < DIM; ++n) {
        float xr = re[n], xi = im[n];
        re[n] = xr * scr - xi * sci;
        im[n] = xr * sci + xi * scr;
    }
    __syncthreads();

    // Emit fragments: frag[(part*2 + tile)*32 + lane]*16 + i, part 0=Re,1=Im,2=-Im
    const int L = threadIdx.x;
    const int chi = (L >= 16) ? 8 : 0;
    #pragma unroll
    for (int t = 0; t < 2; ++t) {
        #pragma unroll
        for (int i = 0; i < 16; ++i) {
            int k = (i < 8 ? i : i + 8) + chi;        // K index this lane/slot holds
            int n = (L & 15) + 16 * t;                // N column
            float mre = sRe[k][n];
            float mim = sIm[k][n];
            frag[((0 * 2 + t) * 32 + L) * 16 + i] = (_Float16)mre;
            frag[((1 * 2 + t) * 32 + L) * 16 + i] = (_Float16)mim;
            frag[((2 * 2 + t) * 32 + L) * 16 + i] = (_Float16)(-mim);
        }
    }
}

// 8 waves/block, 16 batch rows per wave. Complex GEMM row-block via
// v_wmma_f32_16x16x32_f16, then observables through LDS.
__global__ __launch_bounds__(256) void pqc_kernel(const float* __restrict__ noise,
                                                  const _Float16* __restrict__ frag,
                                                  float* __restrict__ out, int B) {
    __shared__ float lds[8][2][16][33];  // [wave][re/im][row M][n] (+pad)
    const int wv   = threadIdx.x >> 5;
    const int lane = threadIdx.x & 31;
    const int b0   = blockIdx.x * 128 + wv * 16;

    // ---- A fragment: noise phases e^{-i t_c} for this lane's 16 K slots ----
    int myb = b0 + (lane & 15);
    int mybc = myb < B ? myb : (B - 1);
    float nz[NQ];
    #pragma unroll
    for (int q = 0; q < NQ; ++q) nz[q] = noise[q * B + mybc];

    const int chi = (lane >> 4) << 3;
    v16h a_re, a_im;
    #pragma unroll
    for (int i = 0; i < 16; ++i) {
        int cidx = (i < 8 ? i : i + 8) + chi;
        float t = 0.f;
        #pragma unroll
        for (int q = 0; q < NQ; ++q)
            t += ((cidx >> (4 - q)) & 1) ? -nz[q] : nz[q];
        t *= 0.5f;
        float sn, cs;
        __sincosf(t, &sn, &cs);
        a_re[i] = (_Float16)cs;
        a_im[i] = (_Float16)(-sn);
    }

    // ---- B fragments (constant, pre-swizzled by build_unitary) ----
    v16h bre0 = *(const v16h*)(frag + (0 * 32 + lane) * 16);
    v16h bre1 = *(const v16h*)(frag + (1 * 32 + lane) * 16);
    v16h bim0 = *(const v16h*)(frag + (2 * 32 + lane) * 16);
    v16h bim1 = *(const v16h*)(frag + (3 * 32 + lane) * 16);
    v16h bni0 = *(const v16h*)(frag + (4 * 32 + lane) * 16);
    v16h bni1 = *(const v16h*)(frag + (5 * 32 + lane) * 16);

    // ---- complex GEMM: S_re = Are*Bre - Aim*Bim ; S_im = Are*Bim + Aim*Bre ----
    v8f cre0 = {}, cre1 = {}, cim0 = {}, cim1 = {};
    cre0 = __builtin_amdgcn_wmma_f32_16x16x32_f16(false, a_re, false, bre0, (short)0, cre0, false, false);
    cre0 = __builtin_amdgcn_wmma_f32_16x16x32_f16(false, a_im, false, bni0, (short)0, cre0, false, false);
    cre1 = __builtin_amdgcn_wmma_f32_16x16x32_f16(false, a_re, false, bre1, (short)0, cre1, false, false);
    cre1 = __builtin_amdgcn_wmma_f32_16x16x32_f16(false, a_im, false, bni1, (short)0, cre1, false, false);
    cim0 = __builtin_amdgcn_wmma_f32_16x16x32_f16(false, a_re, false, bim0, (short)0, cim0, false, false);
    cim0 = __builtin_amdgcn_wmma_f32_16x16x32_f16(false, a_im, false, bre0, (short)0, cim0, false, false);
    cim1 = __builtin_amdgcn_wmma_f32_16x16x32_f16(false, a_re, false, bim1, (short)0, cim1, false, false);
    cim1 = __builtin_amdgcn_wmma_f32_16x16x32_f16(false, a_im, false, bre1, (short)0, cim1, false, false);

    // ---- spill C tiles: VGPR j -> (M = j + 8*(lane>=16), N = lane&15 (+16)) ----
    const int row = (lane >> 4) * 8;
    const int col = lane & 15;
    #pragma unroll
    for (int j = 0; j < 8; ++j) {
        lds[wv][0][row + j][col]      = cre0[j];
        lds[wv][0][row + j][col + 16] = cre1[j];
        lds[wv][1][row + j][col]      = cim0[j];
        lds[wv][1][row + j][col + 16] = cim1[j];
    }
    __syncthreads();

    // ---- observables: x_q = sum_n Re(conj(s_n) s_{n^m}), z_q = probs @ ZS[q] ----
    if (lane < 16) {
        float sr[32], si[32], pp[32];
        #pragma unroll
        for (int n = 0; n < 32; ++n) {
            sr[n] = lds[wv][0][lane][n];
            si[n] = lds[wv][1][lane][n];
            pp[n] = sr[n] * sr[n] + si[n] * si[n];
        }
        int b = b0 + lane;
        float o[10];
        #pragma unroll
        for (int q = 0; q < NQ; ++q) {
            int m = 1 << (4 - q);
            float x = 0.f, z = 0.f;
            #pragma unroll
            for (int n = 0; n < 32; ++n) {
                x += sr[n] * sr[n ^ m] + si[n] * si[n ^ m];
                z += ((n >> (4 - q)) & 1) ? -pp[n] : pp[n];
            }
            o[2 * q]     = x;
            o[2 * q + 1] = z;
        }
        if (b < B) {
            #pragma unroll
            for (int j = 0; j < 10; ++j) out[b * 10 + j] = o[j];
        }
    }
}

extern "C" void kernel_launch(void* const* d_in, const int* in_sizes, int n_in,
                              void* d_out, int out_size, void* d_ws, size_t ws_size,
                              hipStream_t stream) {
    const float* noise = (const float*)d_in[0];   // (5, B) float32
    const float* p     = (const float*)d_in[1];   // (75,) float32
    float* out         = (float*)d_out;           // (B, 10) float32
    _Float16* frag     = (_Float16*)d_ws;         // 6*512 f16 = 6144 bytes

    int B = in_sizes[0] / NQ;

    build_unitary<<<1, 32, 0, stream>>>(p, frag);

    int blocks = (B + 127) / 128;                 // 128 batch rows per block
    pqc_kernel<<<blocks, 256, 0, stream>>>(noise, frag, out, B);
}
